// NeuralJosephsonSDE_31275951849814
// MI455X (gfx1250) — matrix operand order repaired and checked by reading, not compiled, not run on hardware
//
#include <hip/hip_runtime.h>

typedef unsigned short u16b;
typedef unsigned int   u32;
typedef __attribute__((ext_vector_type(16))) __bf16 v16bf;
typedef __attribute__((ext_vector_type(2)))  __bf16 v2bf;
typedef __attribute__((ext_vector_type(8)))  float  v8f;
typedef __attribute__((ext_vector_type(4)))  float  f32x4;

#define NSAMP   32768
#define TSTEPS  512
#define RS      72            // padded LDS row stride (bf16 elems): 144B, bank-conflict-free
#define TBUF    8             // trajectory steps staged in LDS before a coalesced flush
#define STGS    (TBUF*4 + 4)  // staging row stride in floats (36 -> 144B, conflict-free)
#define NETROWS 208           // W1t(64) + W2t(64) + W3t(64) + W4t(16) rows per net
#define BLK     256

// ---------- scalar helpers ----------
__device__ __forceinline__ u16b f2bf(float f){
  union { float f; u32 u; } v; v.f = f;
  u32 u = v.u;
  u += 0x7fffu + ((u >> 16) & 1u);          // round-to-nearest-even
  return (u16b)(u >> 16);
}
__device__ __forceinline__ u32 pack2(float a, float b){
#if __has_builtin(__builtin_amdgcn_cvt_pk_bf16_f32)
  union { v2bf v; u32 u; } c;
  c.v = __builtin_amdgcn_cvt_pk_bf16_f32(a, b);
  return c.u;
#else
  // native casts: let the backend pick v_cvt_*bf16* if gfx1250 has it
  union { v2bf v; u32 u; } c;
  c.v = v2bf{(__bf16)a, (__bf16)b};
  return c.u;
#endif
}
__device__ __forceinline__ float fast_tanh(float x){
#if __has_builtin(__builtin_amdgcn_tanhf)
  return __builtin_amdgcn_tanhf(x);
#else
  float e = __builtin_amdgcn_exp2f(x * -2.885390082f);       // e^{-2x}
  return (1.0f - e) * __builtin_amdgcn_rcpf(1.0f + e);
#endif
}
__device__ __forceinline__ float lipswish(float x){
  // 0.909*x*sigmoid(x) = 0.4545*x*(1 + tanh(x/2))
  float b = 0.4545f * x;
  return fmaf(b, fast_tanh(0.5f * x), b);
}
__device__ __forceinline__ float softplus_stable(float x){
  float e = __builtin_amdgcn_exp2f(-fabsf(x) * 1.4426950408889634f); // e^{-|x|}
  return fmaxf(x, 0.0f) + 0.6931471805599453f * __builtin_amdgcn_logf(1.0f + e);
}

// ---------- WMMA + fragment loaders ----------
__device__ __forceinline__ v8f wmma_bf16(v16bf a, v16bf b, v8f c){
  return __builtin_amdgcn_wmma_f32_16x16x32_bf16(false, a, false, b, (short)0, c, false, false);
}
// A-fragment (16x32, row j = lane&15; low lanes K={c..c+7,c+16..c+23} with c=kc, high lanes c=kc+8)
__device__ __forceinline__ v16bf loadA(const u16b* W, int tj, int kc, int hi, int r15){
  union { v16bf v; f32x4 q[2]; } u;
  const u16b* p = W + (tj*16 + r15)*RS + kc + hi*8;
  u.q[0] = *(const f32x4*)p;
  u.q[1] = *(const f32x4*)(p + 16);
  return u.v;
}
// B-fragment (32x16, col s = lane&15; low lanes K={c..c+7,c+16..c+23} with c=kc, high lanes c=kc+8)
__device__ __forceinline__ v16bf loadB(const u16b* hb, int kc, int hi, int r15){
  union { v16bf v; f32x4 q[2]; } u;
  const u16b* p = hb + r15*RS + kc + hi*8;
  u.q[0] = *(const f32x4*)p;
  u.q[1] = *(const f32x4*)(p + 16);
  return u.v;
}
// C seed = bias[tile*16 + hi*8 + r] : contiguous 32B broadcast read
__device__ __forceinline__ v8f loadC(const float* p){
  union { v8f v; f32x4 q[2]; } u;
  u.q[0] = *(const f32x4*)p;
  u.q[1] = *(const f32x4*)(p + 4);
  return u.v;
}
// D tile (lane s holds rows j = tj*16 + hi*8 + 0..7) -> lipswish -> packed bf16 row store
__device__ __forceinline__ void act_store(const v8f acc[4], u16b* hb, int hi, int r15){
  #pragma unroll
  for (int tj = 0; tj < 4; ++tj){
    union { f32x4 q; u32 u[4]; } s;
    #pragma unroll
    for (int r = 0; r < 4; ++r)
      s.u[r] = pack2(lipswish(acc[tj][2*r]), lipswish(acc[tj][2*r+1]));
    *(f32x4*)(hb + r15*RS + tj*16 + hi*8) = s.q;
  }
}

// full 4-layer MLP: D = W^T x h^T per layer, activations round-trip through hb (per-wave LDS)
// Layer-1 bias is folded into W1t column K=4 (yB carries 1.0 at K=4), so C1 = 0.
__device__ __forceinline__ v8f run_net(const u16b* Wn, const float* Bn, u16b* hb,
                                       v16bf yB, int hi, int r15){
  const u16b* W1 = Wn;
  const u16b* W2 = Wn + 64*RS;
  const u16b* W3 = Wn + 128*RS;
  const u16b* W4 = Wn + 192*RS;
  const v8f zc = {0.f,0.f,0.f,0.f,0.f,0.f,0.f,0.f};
  v8f acc[4];
  // layer 1 (K=4 real + K=4 bias column, padded to 32; W1t cols 5..31 are zero in LDS)
  #pragma unroll
  for (int tj = 0; tj < 4; ++tj)
    acc[tj] = wmma_bf16(loadA(W1, tj, 0, hi, r15), yB, zc);
  act_store(acc, hb, hi, r15);
  // layer 2 (K=64)
  {
    v16bf b0 = loadB(hb, 0, hi, r15);
    v16bf b1 = loadB(hb, 32, hi, r15);
    #pragma unroll
    for (int tj = 0; tj < 4; ++tj){
      v8f c = loadC(Bn + 64 + tj*16 + hi*8);
      c = wmma_bf16(loadA(W2, tj, 0,  hi, r15), b0, c);
      acc[tj] = wmma_bf16(loadA(W2, tj, 32, hi, r15), b1, c);
    }
    act_store(acc, hb, hi, r15);
  }
  // layer 3 (K=64)
  {
    v16bf b0 = loadB(hb, 0, hi, r15);
    v16bf b1 = loadB(hb, 32, hi, r15);
    #pragma unroll
    for (int tj = 0; tj < 4; ++tj){
      v8f c = loadC(Bn + 128 + tj*16 + hi*8);
      c = wmma_bf16(loadA(W3, tj, 0,  hi, r15), b0, c);
      acc[tj] = wmma_bf16(loadA(W3, tj, 32, hi, r15), b1, c);
    }
    act_store(acc, hb, hi, r15);
  }
  // layer 4 (J=4 padded to 16: W4t rows 4..15 and b4 pad are zero)
  v16bf b0 = loadB(hb, 0, hi, r15);
  v16bf b1 = loadB(hb, 32, hi, r15);
  v8f o = loadC(Bn + 192 + hi*8);
  o = wmma_bf16(loadA(W4, 0, 0,  hi, r15), b0, o);
  o = wmma_bf16(loadA(W4, 0, 32, hi, r15), b1, o);
  return o;    // lanes 0..15: elements 0..3 = output dims for sample (lane&15)
}

__global__ __launch_bounds__(BLK, 4) void sde_kernel(
    const float* __restrict__ init, const float* __restrict__ times, const float* __restrict__ noise,
    const float* __restrict__ fW1, const float* __restrict__ fb1,
    const float* __restrict__ fW2, const float* __restrict__ fb2,
    const float* __restrict__ fW3, const float* __restrict__ fb3,
    const float* __restrict__ fW4, const float* __restrict__ fb4,
    const float* __restrict__ gW1, const float* __restrict__ gb1,
    const float* __restrict__ gW2, const float* __restrict__ gb2,
    const float* __restrict__ gW3, const float* __restrict__ gb3,
    const float* __restrict__ gW4, const float* __restrict__ gb4,
    float* __restrict__ out)
{
  __shared__ u16b  sW[2*NETROWS*RS];        // W^T in bf16, padded rows   (~60KB)
  __shared__ float sB[2*208];               // biases (b4 zero-padded to 16)
  __shared__ float sDT[2*(TSTEPS-1)];       // dt and sqrt(dt)
  __shared__ u16b  sH[8*16*RS];             // per-wave activation buffers (~18KB)
  __shared__ float sStg[8*16*STGS];         // per-wave trajectory staging (~18KB)

  const int tid = threadIdx.x;

  // ---- init: zero then fill weight/bias LDS ----
  for (int i = tid; i < 2*NETROWS*RS; i += BLK) sW[i] = 0;
  for (int i = tid; i < 2*208;        i += BLK) sB[i] = 0.0f;
  __syncthreads();

  const float* Wsrc[8] = {fW1, fW2, fW3, fW4, gW1, gW2, gW3, gW4};
  const float* Bsrc[8] = {fb1, fb2, fb3, fb4, gb1, gb2, gb3, gb4};
  #pragma unroll
  for (int n = 0; n < 2; ++n){
    u16b* wb = sW + n*NETROWS*RS;
    for (int i = tid; i < 4*64;  i += BLK){ int k = i>>6, j = i&63; wb[j*RS + k]           = f2bf(Wsrc[n*4+0][i]); }
    for (int i = tid; i < 64;    i += BLK){ wb[i*RS + 4] = f2bf(Bsrc[n*4+0][i]); }   // b1 folded at K=4
    for (int i = tid; i < 64*64; i += BLK){ int k = i>>6, j = i&63; wb[64*RS  + j*RS + k]  = f2bf(Wsrc[n*4+1][i]); }
    for (int i = tid; i < 64*64; i += BLK){ int k = i>>6, j = i&63; wb[128*RS + j*RS + k]  = f2bf(Wsrc[n*4+2][i]); }
    for (int i = tid; i < 64*4;  i += BLK){ int k = i>>2, j = i&3;  wb[192*RS + j*RS + k]  = f2bf(Wsrc[n*4+3][i]); }
    float* bb = sB + n*208;
    for (int i = tid; i < 64; i += BLK){
      bb[64+i] = Bsrc[n*4+1][i]; bb[128+i] = Bsrc[n*4+2][i];
    }
    for (int i = tid; i < 4; i += BLK) bb[192+i] = Bsrc[n*4+3][i];
  }
  for (int i = tid; i < TSTEPS-1; i += BLK){
    float dt = times[i+1] - times[i];
    sDT[i] = dt; sDT[TSTEPS-1 + i] = sqrtf(dt);
  }
  __syncthreads();

  // ---- per-wave persistent state ----
  const int wid  = tid >> 5;
  const int lane = tid & 31;
  const int r15  = lane & 15;
  const int hi   = lane >> 4;
  u16b*  hb  = sH   + wid*(16*RS);
  float* stg = sStg + wid*(16*STGS);
  const int gbase = blockIdx.x*128 + wid*16;
  const int gs    = gbase + r15;            // high lanes mirror low-lane sample (loads dup)

  f32x4 y = *(const f32x4*)(init + (size_t)gs*4);
  if (hi == 0) *(f32x4*)(stg + r15*STGS) = y;                 // stage initial state (slot 0)

  f32x4 dw = __builtin_nontemporal_load((const f32x4*)(noise + (size_t)gs*4)); // t=0

  for (int t = 0; t < TSTEPS-1; ++t){
    // Compiler-level memory barrier: keep weight/bias fragment loads inside the
    // loop as per-step ds_loads instead of hoisting loop-invariant fragments
    // into hundreds of VGPRs (which spilled to scratch in revision 1).
    asm volatile("" ::: "memory");

    const float dt  = sDT[t];
    const float sdt = sDT[TSTEPS-1 + t];

    f32x4 dwn = dw;                                           // software-pipelined noise
    if (t + 1 < TSTEPS-1)
      dwn = __builtin_nontemporal_load((const f32x4*)(noise + ((size_t)(t+1)*NSAMP + gs)*4));

    // y as a B-fragment: lane s<16 holds K=0..3 = y[s][0..3], K=4 = 1.0 (bias input)
    union { v16bf v; u32 u[8]; } yb;
    #pragma unroll
    for (int i = 0; i < 8; ++i) yb.u[i] = 0u;
    yb.u[0] = (hi == 0) ? pack2(y[0], y[1]) : 0u;
    yb.u[1] = (hi == 0) ? pack2(y[2], y[3]) : 0u;
    yb.u[2] = (hi == 0) ? 0x00003f80u : 0u;                   // bf16(1.0) at K=4

    v8f fo = run_net(sW,              sB,       hb, yb.v, hi, r15);  // drift net
    v8f go = run_net(sW + NETROWS*RS, sB + 208, hb, yb.v, hi, r15);  // diffusion net

    #pragma unroll
    for (int d = 0; d < 4; ++d){
      float dr = fminf(fmaxf(fo[d], -100.0f), 100.0f);
      float df = fmaxf(softplus_stable(go[d]), 1e-4f);
      y[d] = fmaf(df * sdt, dw[d], fmaf(dr, dt, y[d]));       // Euler-Maruyama
    }

    const int slot = (t + 1) & (TBUF - 1);
    if (hi == 0) *(f32x4*)(stg + r15*STGS + slot*4) = y;
    if (slot == TBUF - 1){                                    // coalesced flush of 8 steps
      const int chunk = (t + 1) >> 3;
      const int sam = lane >> 1, half = lane & 1;
      const float* src = stg + sam*STGS + half*16;
      float* dst = out + ((size_t)(gbase + sam)*TSTEPS + (size_t)chunk*TBUF)*4 + half*16;
      #pragma unroll
      for (int i = 0; i < 4; ++i){
        f32x4 v = *(const f32x4*)(src + i*4);
        __builtin_nontemporal_store(v, (f32x4*)(dst + i*4));  // write-once output
      }
    }
    dw = dwn;
  }
}

extern "C" void kernel_launch(void* const* d_in, const int* in_sizes, int n_in,
                              void* d_out, int out_size, void* d_ws, size_t ws_size,
                              hipStream_t stream) {
  (void)in_sizes; (void)n_in; (void)out_size; (void)d_ws; (void)ws_size;
  const float* init  = (const float*)d_in[0];
  const float* times = (const float*)d_in[1];
  const float* noise = (const float*)d_in[2];
  const float* fW1 = (const float*)d_in[3],  *fb1 = (const float*)d_in[4];
  const float* fW2 = (const float*)d_in[5],  *fb2 = (const float*)d_in[6];
  const float* fW3 = (const float*)d_in[7],  *fb3 = (const float*)d_in[8];
  const float* fW4 = (const float*)d_in[9],  *fb4 = (const float*)d_in[10];
  const float* gW1 = (const float*)d_in[11], *gb1 = (const float*)d_in[12];
  const float* gW2 = (const float*)d_in[13], *gb2 = (const float*)d_in[14];
  const float* gW3 = (const float*)d_in[15], *gb3 = (const float*)d_in[16];
  const float* gW4 = (const float*)d_in[17], *gb4 = (const float*)d_in[18];

  dim3 grid(NSAMP / 128);   // 8 waves/block * 16 samples/wave
  dim3 block(BLK);
  sde_kernel<<<grid, block, 0, stream>>>(init, times, noise,
      fW1, fb1, fW2, fb2, fW3, fb3, fW4, fb4,
      gW1, gb1, gW2, gb2, gW3, gb3, gW4, gb4,
      (float*)d_out);
}
